// scaled_dot_product_attention_86423331930792
// MI455X (gfx1250) — matrix operand order repaired and checked by reading, not compiled, not run on hardware
//
#include <hip/hip_runtime.h>

// ---------------------------------------------------------------------------
// Flash attention (no scale) + deterministic pre-softmax dropout (p=0.5)
// for MI455X / gfx1250: bf16 WMMA (v_wmma_f32_16x16x32_bf16), f32 accumulate.
// B=4, H=16, S=1024, D=64.
// Pipelined: next KV tile staged in registers while current tile computes.
// ---------------------------------------------------------------------------

typedef __attribute__((ext_vector_type(16))) __bf16        v16bf;
typedef __attribute__((ext_vector_type(8)))  float         v8f;
typedef __attribute__((ext_vector_type(8)))  unsigned int  v8u;

#define S_LEN      1024
#define DHEAD      64
#define QBLK       128          // q rows per workgroup (8 waves x 16)
#define KBLK       64           // kv rows per tile
#define NTILES     (S_LEN / KBLK)
#define NWAVES     8
#define LDS_STRIDE 72           // halfwords per row: 144B, 16B-aligned, non-pow2

__device__ __forceinline__ unsigned short f2bf(float f) {
  unsigned int u = __builtin_bit_cast(unsigned int, f);
  u += 0x7FFFu + ((u >> 16) & 1u);          // round-to-nearest-even
  return (unsigned short)(u >> 16);
}
__device__ __forceinline__ unsigned int packbf(float a, float b) {
  return (unsigned int)f2bf(a) | ((unsigned int)f2bf(b) << 16);
}
__device__ __forceinline__ unsigned int hash32(unsigned int x) {
  x ^= x >> 16; x *= 0x7feb352dU;
  x ^= x >> 15; x *= 0x846ca68bU;
  x ^= x >> 16;
  return x;
}

// B-fragment (32x16 bf16 operand): lane = N + 16*Khalf, p offset by 8*Khalf;
// 8 K values at +0 (K 8h..8h+7) and 8 at +16 (K 16+8h..16+8h+7).
__device__ __forceinline__ v16bf load_b_frag(const unsigned short* p) {
  uint4 u0 = *(const uint4*)p;
  uint4 u1 = *(const uint4*)(p + 16);
  v8u bu = {u0.x, u0.y, u0.z, u0.w, u1.x, u1.y, u1.z, u1.w};
  return __builtin_bit_cast(v16bf, bu);
}
// A-fragment (16x32 bf16) from LDS row p (already offset to k-chunk base):
// 8 halves at +0 (K base..base+7) and 8 at +16 (K base+16..base+23).
__device__ __forceinline__ v16bf load_a_frag_lds(const unsigned short* p) {
  uint4 u0 = *(const uint4*)p;
  uint4 u1 = *(const uint4*)(p + 16);
  v8u au = {u0.x, u0.y, u0.z, u0.w, u1.x, u1.y, u1.z, u1.w};
  return __builtin_bit_cast(v16bf, au);
}

__global__ __launch_bounds__(256) void fa_dropout_kernel(
    const float* __restrict__ Q, const float* __restrict__ K,
    const float* __restrict__ V, float* __restrict__ O) {
  __shared__ __align__(16) unsigned short Kt[KBLK * LDS_STRIDE];          // [kv][d]
  __shared__ __align__(16) unsigned short Vt[DHEAD * LDS_STRIDE];         // [d][kv]
  __shared__ __align__(16) unsigned short Pt[NWAVES][16 * LDS_STRIDE];    // per-wave P

  const int tid  = threadIdx.x;
  const int wave = tid >> 5;
  const int lane = tid & 31;
  const int nlo  = lane & 15;   // N / column lane within half
  const int hi   = lane >> 4;   // half-wave select

  const int bh   = blockIdx.x >> 3;   // b*H + h  (64 total)
  const int qblk = blockIdx.x & 7;    // q block of 128
  const int q0   = qblk * QBLK + wave * 16;   // this wave's 16 q rows

  const float* Qb = Q + (size_t)bh * S_LEN * DHEAD;
  const float* Kb = K + (size_t)bh * S_LEN * DHEAD;
  const float* Vb = V + (size_t)bh * S_LEN * DHEAD;
  float*       Ob = O + (size_t)bh * S_LEN * DHEAD;

  // cooperative-load geometry: 1024 float4 chunks over 256 threads x 4 steps
  int ldrow[4], ldcv[4];
  #pragma unroll
  for (int i = 0; i < 4; ++i) {
    const int linear = tid + i * 256;
    ldrow[i] = linear >> 4;        // kv row 0..63
    ldcv[i]  = linear & 15;        // float4 index within row
  }

  // ---- Q rows -> two bf16 A-fragments (d 0..31, 32..63), ISA A layout.
  v16bf qa[2];
  {
    const float* qrow = Qb + (size_t)(q0 + nlo) * DHEAD;
    #pragma unroll
    for (int f = 0; f < 2; ++f) {
      v8u au;
      #pragma unroll
      for (int c = 0; c < 2; ++c) {
        const int d0 = f * 32 + hi * 8 + c * 16;
        float4 x = *(const float4*)(qrow + d0);
        float4 y = *(const float4*)(qrow + d0 + 4);
        au[c * 4 + 0] = packbf(x.x, x.y);
        au[c * 4 + 1] = packbf(x.z, x.w);
        au[c * 4 + 2] = packbf(y.x, y.y);
        au[c * 4 + 3] = packbf(y.z, y.w);
      }
      qa[f] = __builtin_bit_cast(v16bf, au);
    }
  }

  v8f acc[4] = {};            // O accumulator: 4 d-tiles of 16x16 f32
  float m_r[8], l_r[8];       // per-row running max / sum (rows v+8*hi)
  #pragma unroll
  for (int v = 0; v < 8; ++v) { m_r[v] = -1e30f; l_r[v] = 0.0f; }

  // row-group id for dropout hashing: this half-wave owns q rows
  // q0+8*hi .. q0+8*hi+7  ->  one hash yields 8 keep bits (bit v = row v)
  const unsigned rg = (unsigned)((q0 >> 3) + hi);   // 0..127 within head

  // ---- stage tile 0 in registers
  float4 kreg[4], vreg[4];
  #pragma unroll
  for (int i = 0; i < 4; ++i) {
    const size_t gofs = (size_t)ldrow[i] * DHEAD + ldcv[i] * 4;
    kreg[i] = *(const float4*)(Kb + gofs);
    vreg[i] = *(const float4*)(Vb + gofs);
  }

  for (int kvb = 0; kvb < NTILES; ++kvb) {
    // ---- commit staged registers to LDS: K -> [kv][d], V -> [d][kv]^T
    #pragma unroll
    for (int i = 0; i < 4; ++i) {
      *(uint2*)&Kt[ldrow[i] * LDS_STRIDE + ldcv[i] * 4] =
          make_uint2(packbf(kreg[i].x, kreg[i].y), packbf(kreg[i].z, kreg[i].w));
      Vt[(ldcv[i] * 4 + 0) * LDS_STRIDE + ldrow[i]] = f2bf(vreg[i].x);
      Vt[(ldcv[i] * 4 + 1) * LDS_STRIDE + ldrow[i]] = f2bf(vreg[i].y);
      Vt[(ldcv[i] * 4 + 2) * LDS_STRIDE + ldrow[i]] = f2bf(vreg[i].z);
      Vt[(ldcv[i] * 4 + 3) * LDS_STRIDE + ldrow[i]] = f2bf(vreg[i].w);
    }
    __syncthreads();

    // ---- issue next tile's global loads now; latency hidden under compute
    if (kvb + 1 < NTILES) {
      #pragma unroll
      for (int i = 0; i < 4; ++i) {
        const size_t gofs =
            (size_t)((kvb + 1) * KBLK + ldrow[i]) * DHEAD + ldcv[i] * 4;
        kreg[i] = *(const float4*)(Kb + gofs);
        vreg[i] = *(const float4*)(Vb + gofs);
      }
    }

    // ---- S = Q * K^T : 4 column tiles of 16x16, K-dim 64 = 2 WMMAs each
    v8f sc[4];
    #pragma unroll
    for (int nt = 0; nt < 4; ++nt) {
      const unsigned short* bp = &Kt[(nt * 16 + nlo) * LDS_STRIDE + hi * 8];
      v16bf kb0 = load_b_frag(bp);        // d 0..31
      v16bf kb1 = load_b_frag(bp + 32);   // d 32..63
      v8f z = {};
      v8f s = __builtin_amdgcn_wmma_f32_16x16x32_bf16(false, qa[0], false, kb0,
                                                      (short)0, z, false, false);
      s = __builtin_amdgcn_wmma_f32_16x16x32_bf16(false, qa[1], false, kb1,
                                                  (short)0, s, false, false);
      sc[nt] = s;
    }

    // ---- deterministic dropout (p=0.5): one hash -> 8 row keep-bits
    unsigned kbits[4];
    #pragma unroll
    for (int nt = 0; nt < 4; ++nt) {
      const unsigned kvg = (unsigned)(kvb * KBLK + nt * 16 + nlo);
      const unsigned idx = ((((unsigned)bh << 7) | rg) << 10) | kvg;
      kbits[nt] = hash32(idx * 0x9E3779B9u + 0x2545F491u);
    }

    // ---- online softmax update (rows live across 16 lanes of each half)
    #pragma unroll
    for (int v = 0; v < 8; ++v) {
      float tmax = -1e30f;
      #pragma unroll
      for (int nt = 0; nt < 4; ++nt) {
        float s = sc[nt][v];
        s = ((kbits[nt] >> v) & 1u) ? s * 2.0f : 0.0f;  // dropped -> zero score
        sc[nt][v] = s;
        tmax = fmaxf(tmax, s);
      }
      #pragma unroll
      for (int off = 1; off < 16; off <<= 1)
        tmax = fmaxf(tmax, __shfl_xor(tmax, off, 16));
      const float mnew  = fmaxf(m_r[v], tmax);
      const float alpha = __expf(m_r[v] - mnew);
      m_r[v] = mnew;
      float rsum = 0.0f;
      #pragma unroll
      for (int nt = 0; nt < 4; ++nt) {
        const float p = __expf(sc[nt][v] - mnew);
        sc[nt][v] = p;
        rsum += p;
      }
      #pragma unroll
      for (int off = 1; off < 16; off <<= 1)
        rsum += __shfl_xor(rsum, off, 16);
      l_r[v] = l_r[v] * alpha + rsum;
      #pragma unroll
      for (int t = 0; t < 4; ++t) acc[t][v] *= alpha;
    }

    // ---- C-layout -> A-layout for P via per-wave LDS scratch (bf16)
    unsigned short* Pw = &Pt[wave][0];
    #pragma unroll
    for (int nt = 0; nt < 4; ++nt)
      #pragma unroll
      for (int v = 0; v < 8; ++v)
        Pw[(v + 8 * hi) * LDS_STRIDE + nt * 16 + nlo] = f2bf(sc[nt][v]);
    asm volatile("s_wait_dscnt 0" ::: "memory");

    v16bf pa[2];
    #pragma unroll
    for (int f = 0; f < 2; ++f)
      pa[f] = load_a_frag_lds(&Pw[nlo * LDS_STRIDE + f * 32 + hi * 8]);

    // ---- O += P * V : 4 d-tiles, K-dim 64 = 2 WMMAs each
    #pragma unroll
    for (int t = 0; t < 4; ++t) {
      const unsigned short* bp = &Vt[(t * 16 + nlo) * LDS_STRIDE + hi * 8];
      v16bf vb0 = load_b_frag(bp);        // kv 0..31
      v16bf vb1 = load_b_frag(bp + 32);   // kv 32..63
      acc[t] = __builtin_amdgcn_wmma_f32_16x16x32_bf16(false, pa[0], false, vb0,
                                                       (short)0, acc[t], false, false);
      acc[t] = __builtin_amdgcn_wmma_f32_16x16x32_bf16(false, pa[1], false, vb1,
                                                       (short)0, acc[t], false, false);
    }
    __syncthreads();   // protect Kt/Vt before next tile's overwrite
  }

  // ---- normalize and store O (f32)
  #pragma unroll
  for (int v = 0; v < 8; ++v) {
    const float inv = 1.0f / l_r[v];
    float* orow = Ob + (size_t)(q0 + v + 8 * hi) * DHEAD;
    #pragma unroll
    for (int t = 0; t < 4; ++t)
      orow[t * 16 + nlo] = acc[t][v] * inv;
  }
}

extern "C" void kernel_launch(void* const* d_in, const int* in_sizes, int n_in,
                              void* d_out, int out_size, void* d_ws, size_t ws_size,
                              hipStream_t stream) {
  (void)in_sizes; (void)n_in; (void)out_size; (void)d_ws; (void)ws_size;
  const float* Q = (const float*)d_in[0];
  const float* K = (const float*)d_in[1];
  const float* V = (const float*)d_in[2];
  float* O = (float*)d_out;
  // grid: 64 (B*H) heads x 8 q-blocks of 128 rows
  fa_dropout_kernel<<<dim3(64 * 8), dim3(256), 0, stream>>>(Q, K, V, O);
}
